// RNNFromCell_37117107372581
// MI455X (gfx1250) — matrix operand order repaired and checked by reading, not compiled, not run on hardware
//
#include <hip/hip_runtime.h>

// ---------------- problem constants ----------------
#define T_SEQ 512
#define NB    64      // batch
#define NH    1024    // hidden == input size
#define NG    4096    // 4*NH gates

// LDS weight strip: 64 rows (4 gates x 16 cols) of 1024 bf16, padded +8 shorts
// (+16B) per row so B-fragment reads stripe across all 64 LDS banks.
#define ROW_SHORTS 1032
#define LDS_SHORTS (2 * 64 * ROW_SHORTS)          // wih strip + whh strip
#define LDS_BYTES  (LDS_SHORTS * 2)               // 264,192 B < 320 KB WGP LDS

typedef __attribute__((ext_vector_type(16))) __bf16 v16bf;
typedef __attribute__((ext_vector_type(8)))  float  v8f;

union V16 { v16bf v; float4 q[2]; };

__device__ __forceinline__ unsigned short f32_to_bf16(float f) {
    unsigned u = __float_as_uint(f);
    unsigned r = 0x7FFFu + ((u >> 16) & 1u);   // round-to-nearest-even
    return (unsigned short)((u + r) >> 16);
}

__device__ __forceinline__ float sigmoid_f(float x) {
    return 1.0f / (1.0f + __expf(-x));
}
__device__ __forceinline__ float tanh_f(float x) {
    float e = __expf(-2.0f * x);
    return (1.0f - e) / (1.0f + e);
}

// GEMM phase over K=1024: A tile [16 x K] from global (lane-resolved row ptr),
// 4 gate B tiles [K x 16] from LDS (lane-resolved column ptrs).
// Per k-step: 2x global_load_b128 (A), 8x ds_load_b128 (B), 4x wmma_bf16.
__device__ __forceinline__ void gemm_phase(
    const unsigned short* __restrict__ aBase,
    const unsigned short* __restrict__ b0, const unsigned short* __restrict__ b1,
    const unsigned short* __restrict__ b2, const unsigned short* __restrict__ b3,
    v8f& acc0, v8f& acc1, v8f& acc2, v8f& acc3)
{
#pragma unroll 4
    for (int k0 = 0; k0 < NH; k0 += 32) {
        V16 a;
        a.q[0] = *(const float4*)(aBase + k0);        // lanes<16: K k0..k0+7  | hi: k0+16..23
        a.q[1] = *(const float4*)(aBase + k0 + 8);    // lanes<16: K k0+8..15  | hi: k0+24..31
        V16 w0, w1, w2, w3;
        w0.q[0] = *(const float4*)(b0 + k0); w0.q[1] = *(const float4*)(b0 + k0 + 8);
        w1.q[0] = *(const float4*)(b1 + k0); w1.q[1] = *(const float4*)(b1 + k0 + 8);
        w2.q[0] = *(const float4*)(b2 + k0); w2.q[1] = *(const float4*)(b2 + k0 + 8);
        w3.q[0] = *(const float4*)(b3 + k0); w3.q[1] = *(const float4*)(b3 + k0 + 8);
        acc0 = __builtin_amdgcn_wmma_f32_16x16x32_bf16(false, a.v, false, w0.v, (short)0, acc0, false, false);
        acc1 = __builtin_amdgcn_wmma_f32_16x16x32_bf16(false, a.v, false, w1.v, (short)0, acc1, false, false);
        acc2 = __builtin_amdgcn_wmma_f32_16x16x32_bf16(false, a.v, false, w2.v, (short)0, acc2, false, false);
        acc3 = __builtin_amdgcn_wmma_f32_16x16x32_bf16(false, a.v, false, w3.v, (short)0, acc3, false, false);
    }
}

// fp32 -> bf16 converter, 4 elements/thread, grid-stride
__global__ void cvt_f32_bf16(const float* __restrict__ src,
                             unsigned short* __restrict__ dst, int n4) {
    int i = blockIdx.x * blockDim.x + threadIdx.x;
    int stride = gridDim.x * blockDim.x;
    const float4* s4 = (const float4*)src;
    ushort4* d4 = (ushort4*)dst;
    for (; i < n4; i += stride) {
        float4 v = s4[i];
        ushort4 o;
        o.x = f32_to_bf16(v.x); o.y = f32_to_bf16(v.y);
        o.z = f32_to_bf16(v.z); o.w = f32_to_bf16(v.w);
        d4[i] = o;
    }
}

__global__ void zero_ctr(unsigned int* c) {
    if (blockIdx.x == 0 && threadIdx.x == 0) *c = 0u;
}

// Persistent LSTM. grid = 64 blocks (hidden strips of 16), block = 128 threads
// (4 waves = 4 batch tiles of 16). Weights live in LDS for all 512 steps;
// c-state lives in VGPRs; split arrive/wait barrier hides the x-GEMM.
__global__ void __launch_bounds__(128, 1)
lstm_persistent(const unsigned short* __restrict__ x_bf,    // [T,B,NH] bf16
                const unsigned short* __restrict__ wih_bf,  // [NG,NH]  bf16
                const unsigned short* __restrict__ whh_bf,  // [NG,NH]  bf16
                const float* __restrict__ b_ih,
                const float* __restrict__ b_hh,
                float* __restrict__ out,                    // all_h|all_c|h_last|c_last
                unsigned short* __restrict__ hbuf,          // [2][B*NH] bf16 ping-pong
                unsigned int* __restrict__ barrier_ctr)
{
    extern __shared__ unsigned short lds_w[];               // [2][64][ROW_SHORTS]
    unsigned short* lds_wih = lds_w;
    unsigned short* lds_whh = lds_w + (size_t)64 * ROW_SHORTS;

    const int lane  = threadIdx.x & 31;
    const int wave  = threadIdx.x >> 5;
    const int m0    = wave * 16;                 // batch tile base
    const int n0    = blockIdx.x * 16;           // hidden strip base
    const int col   = n0 + (lane & 15);          // hidden column of this lane
    const int rowA  = m0 + (lane & 15);          // A-matrix row of this lane
    const int ahalf = (lane >> 4) << 4;          // +0 / +16 (A K-half select)
    const int bhalf = (lane >> 4) << 4;          // +0 / +16 (B K-half select)

    // ---- one-time: stage this block's weight strips into LDS ----
    // 64 rows x 128 chunks (16B) per strip; rows = gate*16 + c, c in [0,16)
    for (int i = threadIdx.x; i < 64 * 128; i += blockDim.x) {
        const int row   = i >> 7;
        const int chunk = i & 127;
        const int gate  = row >> 4;
        const int c     = row & 15;
        const size_t gsrc = (size_t)(gate * NH + n0 + c) * NH;
        const float4 vih = *((const float4*)(wih_bf + gsrc) + chunk);
        const float4 vhh = *((const float4*)(whh_bf + gsrc) + chunk);
        *((float4*)(lds_wih + (size_t)row * ROW_SHORTS) + chunk) = vih;
        *((float4*)(lds_whh + (size_t)row * ROW_SHORTS) + chunk) = vhh;
    }
    __syncthreads();

    // per-lane gate biases (column-constant across the tile)
    const float bias_i = b_ih[col]          + b_hh[col];
    const float bias_f = b_ih[NH + col]     + b_hh[NH + col];
    const float bias_g = b_ih[2 * NH + col] + b_hh[2 * NH + col];
    const float bias_o = b_ih[3 * NH + col] + b_hh[3 * NH + col];

    // lane-resolved LDS B pointers (row = gate*16 + lane%16, +K-half)
    const unsigned short* wi0 = lds_wih + (size_t)( 0 + (lane & 15)) * ROW_SHORTS + bhalf;
    const unsigned short* wi1 = lds_wih + (size_t)(16 + (lane & 15)) * ROW_SHORTS + bhalf;
    const unsigned short* wi2 = lds_wih + (size_t)(32 + (lane & 15)) * ROW_SHORTS + bhalf;
    const unsigned short* wi3 = lds_wih + (size_t)(48 + (lane & 15)) * ROW_SHORTS + bhalf;
    const unsigned short* wh0 = lds_whh + (size_t)( 0 + (lane & 15)) * ROW_SHORTS + bhalf;
    const unsigned short* wh1 = lds_whh + (size_t)(16 + (lane & 15)) * ROW_SHORTS + bhalf;
    const unsigned short* wh2 = lds_whh + (size_t)(32 + (lane & 15)) * ROW_SHORTS + bhalf;
    const unsigned short* wh3 = lds_whh + (size_t)(48 + (lane & 15)) * ROW_SHORTS + bhalf;

    v8f cstate = {};   // c0 = 0, lives in VGPRs for the whole sequence

    float* all_h  = out;
    float* all_c  = out + (size_t)T_SEQ * NB * NH;
    float* h_last = out + 2 * (size_t)T_SEQ * NB * NH;
    float* c_last = h_last + (size_t)NB * NH;

    for (int t = 0; t < T_SEQ; ++t) {
        v8f acc_i, acc_f, acc_g, acc_o;
#pragma unroll
        for (int r = 0; r < 8; ++r) {
            acc_i[r] = bias_i; acc_f[r] = bias_f;
            acc_g[r] = bias_g; acc_o[r] = bias_o;
        }

        // ---- x-phase: gates += x[t] @ W_ih^T  (independent of h_{t-1}) ----
        const unsigned short* aX =
            x_bf + (size_t)t * NB * NH + (size_t)rowA * NH + ahalf;
        gemm_phase(aX, wi0, wi1, wi2, wi3, acc_i, acc_f, acc_g, acc_o);

        // hint: pull next timestep's x rows toward the caches
        if (t + 1 < T_SEQ) {
            __builtin_prefetch(x_bf + (size_t)(t + 1) * NB * NH +
                               (size_t)rowA * NH + (lane >> 4) * 512, 0, 1);
        }

        // ---- wait for h_{t-1}, then h-phase (x-GEMM hid the barrier) ----
        if (t > 0) {
            if (threadIdx.x == 0) {
                const unsigned target = (unsigned)gridDim.x * (unsigned)t;
                while (atomicAdd(barrier_ctr, 0u) < target) {
                    __builtin_amdgcn_s_sleep(2);
                }
            }
            __syncthreads();
            __threadfence();   // acquire: make other blocks' h stores visible
            const unsigned short* aH =
                hbuf + (size_t)((t & 1) ^ 1) * NB * NH + (size_t)rowA * NH + ahalf;
            gemm_phase(aH, wh0, wh1, wh2, wh3, acc_i, acc_f, acc_g, acc_o);
        }

        // ---- pointwise LSTM update, all in registers ----
        unsigned short* hout = hbuf + (size_t)(t & 1) * NB * NH;
        const size_t tbase = (size_t)t * NB * NH;
#pragma unroll
        for (int r = 0; r < 8; ++r) {
            float gi = sigmoid_f(acc_i[r]);
            float gf = sigmoid_f(acc_f[r]);
            float gg = tanh_f(acc_g[r]);
            float go = sigmoid_f(acc_o[r]);
            float c  = gf * cstate[r] + gi * gg;
            float h  = go * tanh_f(c);
            cstate[r] = c;
            const int b   = m0 + r + ((lane >> 4) << 3);   // C/D layout: M = r (+8 hi lanes)
            const int idx = b * NH + col;
            all_h[tbase + idx] = h;
            all_c[tbase + idx] = c;
            hout[idx] = f32_to_bf16(h);
            if (t == T_SEQ - 1) { h_last[idx] = h; c_last[idx] = c; }
        }

        // ---- arrive: signal completion of step t (no wait here) ----
        __threadfence();       // release: h stores visible device-wide
        __syncthreads();
        if (threadIdx.x == 0) atomicAdd(barrier_ctr, 1u);
    }
}

extern "C" void kernel_launch(void* const* d_in, const int* in_sizes, int n_in,
                              void* d_out, int out_size, void* d_ws, size_t ws_size,
                              hipStream_t stream) {
    const float* x    = (const float*)d_in[0];   // [512,64,1024]
    const float* W_ih = (const float*)d_in[1];   // [4096,1024]
    const float* W_hh = (const float*)d_in[2];   // [4096,1024]
    const float* b_ih = (const float*)d_in[3];   // [4096]
    const float* b_hh = (const float*)d_in[4];   // [4096]
    float* out = (float*)d_out;

    // workspace layout
    char* ws = (char*)d_ws;
    unsigned int*   ctr    = (unsigned int*)ws;                      // 256 B slot
    unsigned short* x_bf   = (unsigned short*)(ws + 256);            // 33,554,432 elems
    unsigned short* wih_bf = x_bf   + (size_t)T_SEQ * NB * NH;       //  4,194,304 elems
    unsigned short* whh_bf = wih_bf + (size_t)NG * NH;               //  4,194,304 elems
    unsigned short* hbuf   = whh_bf + (size_t)NG * NH;               //  2 * 65,536 elems

    zero_ctr<<<1, 64, 0, stream>>>(ctr);
    cvt_f32_bf16<<<4096, 256, 0, stream>>>(x,    x_bf,   (T_SEQ * NB * NH) / 4);
    cvt_f32_bf16<<<1024, 256, 0, stream>>>(W_ih, wih_bf, (NG * NH) / 4);
    cvt_f32_bf16<<<1024, 256, 0, stream>>>(W_hh, whh_bf, (NG * NH) / 4);

    lstm_persistent<<<64, 128, LDS_BYTES, stream>>>(x_bf, wih_bf, whh_bf,
                                                    b_ih, b_hh, out, hbuf, ctr);
}
